// NERModel_46669114638828
// MI455X (gfx1250) — compile-verified, matrix operand order checked by reading, not run on hardware
//
#include <hip/hip_runtime.h>
#include <hip/hip_bf16.h>

#define BB 512
#define TT 128
#define EE 128
#define HH 256
#define GG 1024   // 4*HH
#define KT 10
#define KTP 16
#define ROWS 32   // batch rows per LSTM block (2 WMMA M-tiles -> 2x B-reuse)

typedef _Float16 v16h __attribute__((ext_vector_type(16)));
typedef _Float16 v8h  __attribute__((ext_vector_type(8)));
typedef float    v8f  __attribute__((ext_vector_type(8)));

__device__ inline v16h cat16(v8h lo, v8h hi) {
    return __builtin_shufflevector(lo, hi, 0,1,2,3,4,5,6,7,8,9,10,11,12,13,14,15);
}

// activations via v_exp_f32 + v_rcp_f32 (TRANS pipe, co-executes with WMMA);
// avoids IEEE v_div_fmas/v_div_fixup expansion and ocml tanh.
__device__ inline float fast_sigm(float x) {
    return __builtin_amdgcn_rcpf(1.0f + __expf(-x));
}
__device__ inline float fast_tanh(float x) {
    return __builtin_fmaf(2.0f, __builtin_amdgcn_rcpf(1.0f + __expf(-2.0f * x)), -1.0f);
}

// ---------------- embedding gather + f32->f16 ----------------
__global__ __launch_bounds__(256) void k_embed(const int* __restrict__ word,
                                               const float* __restrict__ emb,
                                               _Float16* __restrict__ xh) {
    size_t i = (size_t)blockIdx.x * 256 + threadIdx.x;   // over BB*TT*EE
    int bt = (int)(i >> 7);
    int e  = (int)(i & 127);
    int w  = word[bt];
    xh[i] = (_Float16)emb[(size_t)w * EE + e];
}

// ---------------- lengths ----------------
__global__ __launch_bounds__(256) void k_len(const int* __restrict__ word,
                                             int* __restrict__ len_i,
                                             float* __restrict__ len_f) {
    int b = blockIdx.x * 256 + threadIdx.x;
    if (b < BB) {
        int c = 0;
        for (int t = 0; t < TT; ++t) c += (word[b * TT + t] != 0) ? 1 : 0;
        len_i[b] = c;
        len_f[b] = (float)c;
    }
}

// ---------------- transpose f32[K][N] -> f16[N][K] ----------------
__global__ __launch_bounds__(256) void k_t16(_Float16* __restrict__ dst,
                                             const float* __restrict__ src,
                                             int K, int N) {
    int i = blockIdx.x * 256 + threadIdx.x;  // i = k*N + n
    if (i < K * N) {
        int k = i / N, n = i % N;
        dst[(size_t)n * K + k] = (_Float16)src[i];
    }
}

// ---------------- fused BiLSTM + partial dense ----------------
// grid: 32 blocks (blk>>4 = direction, blk&15 = 32-row batch tile), 512 threads = 16 waves.
// wave w owns columns [w*16, w*16+16) of each gate (4 N-tiles: i,f,g,o) x 2 M-tiles.
// 3 barriers per timestep; next x tile staged during the logits phase.
__global__ __launch_bounds__(512) void k_lstm(
    const _Float16* __restrict__ xh,
    const _Float16* __restrict__ WtF, const _Float16* __restrict__ UtF,
    const _Float16* __restrict__ WtB, const _Float16* __restrict__ UtB,
    const float* __restrict__ bf, const float* __restrict__ bb,
    const float* __restrict__ Wd,
    float* __restrict__ lgf, float* __restrict__ lgb)
{
    __shared__ __align__(16) _Float16 h_lds[ROWS][HH + 8];   // current hidden (f16)
    __shared__ __align__(16) _Float16 x_lds[ROWS][EE + 8];   // staged x_t tile
    __shared__ __align__(16) _Float16 wdt_lds[16][HH + 8];   // Wd slice, transposed, tag-padded
    __shared__ float lg_part[16][256];                       // per-wave logit partials

    const int tid  = threadIdx.x;
    const int wave = tid >> 5;
    const int lane = tid & 31;
    const int n16  = lane & 15;       // N within tile / A row
    const int m    = lane & 15;
    const int hi   = lane >> 4;       // half-wave select
    const int co   = hi << 3;         // K chunk offset (0 or 8)

    const int dir  = blockIdx.x >> 4;
    const int row0 = (blockIdx.x & 15) << 5;

    const _Float16* Wt   = dir ? WtB : WtF;
    const _Float16* Ut   = dir ? UtB : UtF;
    const float*    bias = dir ? bb  : bf;
    float*          lg   = dir ? lgb : lgf;

    // staging coordinates (all 512 threads move one 16B chunk)
    const int sr = tid >> 4;          // row 0..31
    const int sc = (tid & 15) << 3;   // halfword column chunk

    // init: zero h, build transposed f16 Wd slice (pad tags 10..15 with 0)
    for (int i = tid; i < ROWS * (HH + 8); i += 512) (&h_lds[0][0])[i] = (_Float16)0.0f;
    for (int i = tid; i < 16 * HH; i += 512) {
        int n = i & 15, k = i >> 4;
        wdt_lds[n][k] = (n < KT) ? (_Float16)Wd[(dir * HH + k) * KT + n] : (_Float16)0.0f;
    }

    // per-lane bias and N-tile bases: gate g, columns g*256 + wave*16 + n16
    float bv[4];
    int   nb[4];
    #pragma unroll
    for (int g = 0; g < 4; ++g) {
        nb[g] = (g << 8) + (wave << 4);
        bv[g] = bias[nb[g] + n16];
    }

    v8f cst[2];                        // cell state (f32), C-tile layout, per M-tile
    #pragma unroll
    for (int mt = 0; mt < 2; ++mt)
        #pragma unroll
        for (int r = 0; r < 8; ++r) cst[mt][r] = 0.0f;

    // prologue: stage x tile for step 0
    {
        const int t0 = dir ? (TT - 1) : 0;
        const _Float16* src = xh + (((size_t)(row0 + sr)) * TT + t0) * EE + sc;
        *(v8h*)&x_lds[sr][sc] = *(const v8h*)src;
    }
    __syncthreads();

    for (int step = 0; step < TT; ++step) {
        const int t = dir ? (TT - 1 - step) : step;

        // phase B: z = x_t @ W + h @ U  (f16 WMMA, f32 accum), 2 M-tiles reuse each B tile
        v8f acc[2][4];
        #pragma unroll
        for (int mt = 0; mt < 2; ++mt)
            #pragma unroll
            for (int g = 0; g < 4; ++g)
                #pragma unroll
                for (int r = 0; r < 8; ++r) acc[mt][g][r] = 0.0f;

        for (int kb = 0; kb < EE; kb += 32) {
            v16h a0 = cat16(*(const v8h*)&x_lds[m][kb + co],
                            *(const v8h*)&x_lds[m][kb + co + 16]);
            v16h a1 = cat16(*(const v8h*)&x_lds[16 + m][kb + co],
                            *(const v8h*)&x_lds[16 + m][kb + co + 16]);
            #pragma unroll
            for (int g = 0; g < 4; ++g) {
                const _Float16* bp = Wt + (size_t)(nb[g] + n16) * EE + kb + co;
                v16h bm = cat16(*(const v8h*)bp, *(const v8h*)(bp + 16));
                acc[0][g] = __builtin_amdgcn_wmma_f32_16x16x32_f16(
                    false, a0, false, bm, (short)0, acc[0][g], false, false);
                acc[1][g] = __builtin_amdgcn_wmma_f32_16x16x32_f16(
                    false, a1, false, bm, (short)0, acc[1][g], false, false);
            }
        }
        for (int kb = 0; kb < HH; kb += 32) {
            v16h a0 = cat16(*(const v8h*)&h_lds[m][kb + co],
                            *(const v8h*)&h_lds[m][kb + co + 16]);
            v16h a1 = cat16(*(const v8h*)&h_lds[16 + m][kb + co],
                            *(const v8h*)&h_lds[16 + m][kb + co + 16]);
            #pragma unroll
            for (int g = 0; g < 4; ++g) {
                const _Float16* bp = Ut + (size_t)(nb[g] + n16) * HH + kb + co;
                v16h bm = cat16(*(const v8h*)bp, *(const v8h*)(bp + 16));
                acc[0][g] = __builtin_amdgcn_wmma_f32_16x16x32_f16(
                    false, a0, false, bm, (short)0, acc[0][g], false, false);
                acc[1][g] = __builtin_amdgcn_wmma_f32_16x16x32_f16(
                    false, a1, false, bm, (short)0, acc[1][g], false, false);
            }
        }
        __syncthreads();   // sync1: all reads of old h and current x complete

        // phase C: gates, state update, write new h (f16) to LDS
        #pragma unroll
        for (int mt = 0; mt < 2; ++mt) {
            #pragma unroll
            for (int r = 0; r < 8; ++r) {
                float iv = fast_sigm(acc[mt][0][r] + bv[0]);
                float fv = fast_sigm(acc[mt][1][r] + bv[1]);
                float gv = fast_tanh(acc[mt][2][r] + bv[2]);
                float ov = fast_sigm(acc[mt][3][r] + bv[3]);
                float cv = fv * cst[mt][r] + iv * gv;
                cst[mt][r] = cv;
                float hv = ov * fast_tanh(cv);
                int Mrow = (mt << 4) + r + (hi << 3);
                h_lds[Mrow][(wave << 4) + n16] = (_Float16)hv;
            }
        }
        __syncthreads();   // sync2: new h visible, previous lg_part consumed

        // phase D: fused dense partial (h[32x256] @ WdSlice[256x16]) — 16 wave-tasks:
        // wave w: M-tile w>>3, K-slice (w&7)*32. Also stage next x tile (overlaps WMMA).
        {
            int mt = wave >> 3;
            int ks = (wave & 7) << 5;
            v16h a  = cat16(*(const v8h*)&h_lds[(mt << 4) + m][ks + co],
                            *(const v8h*)&h_lds[(mt << 4) + m][ks + co + 16]);
            v16h bm = cat16(*(const v8h*)&wdt_lds[n16][ks + co],
                            *(const v8h*)&wdt_lds[n16][ks + co + 16]);
            v8f z;
            #pragma unroll
            for (int r = 0; r < 8; ++r) z[r] = 0.0f;
            z = __builtin_amdgcn_wmma_f32_16x16x32_f16(
                false, a, false, bm, (short)0, z, false, false);
            #pragma unroll
            for (int r = 0; r < 8; ++r) {
                int Mrow = r + (hi << 3);
                lg_part[wave][Mrow * 16 + n16] = z[r];
            }
            if (step + 1 < TT) {
                const int tn = dir ? (TT - 2 - step) : (step + 1);
                const _Float16* src = xh + (((size_t)(row0 + sr)) * TT + tn) * EE + sc;
                *(v8h*)&x_lds[sr][sc] = *(const v8h*)src;
            }
        }
        __syncthreads();   // sync3: partials ready, next x staged

        // phase E: tree-sum K-slice partials, store partial logits for this direction
        {
            int r = tid >> 4, n = tid & 15;       // r in 0..31
            int mt = r >> 4, rr = r & 15;
            float s = 0.0f;
            #pragma unroll
            for (int j = 0; j < 8; ++j) s += lg_part[(mt << 3) + j][rr * 16 + n];
            lg[(((size_t)(row0 + r)) * TT + t) * KTP + n] = s;
        }
    }
}

// ---------------- dense bias + softmax ----------------
__global__ __launch_bounds__(256) void k_soft(const float* __restrict__ lgf,
                                              const float* __restrict__ lgb,
                                              const float* __restrict__ bd,
                                              float* __restrict__ probs) {
    int bt = blockIdx.x * 256 + threadIdx.x;
    if (bt < BB * TT) {
        float v[KT];
        float mx = -1e30f;
        #pragma unroll
        for (int k = 0; k < KT; ++k) {
            v[k] = lgf[(size_t)bt * KTP + k] + lgb[(size_t)bt * KTP + k] + bd[k];
            mx = fmaxf(mx, v[k]);
        }
        float s = 0.0f;
        #pragma unroll
        for (int k = 0; k < KT; ++k) { v[k] = __expf(v[k] - mx); s += v[k]; }
        float inv = __builtin_amdgcn_rcpf(s);
        #pragma unroll
        for (int k = 0; k < KT; ++k) probs[(size_t)bt * KT + k] = v[k] * inv;
    }
}

// ---------------- CRF log-likelihood: one wave32 per sequence ----------------
__global__ __launch_bounds__(256) void k_crf(const float* __restrict__ probs,
                                             const int* __restrict__ label,
                                             const int* __restrict__ len_i,
                                             const float* __restrict__ trans,
                                             float* __restrict__ out_ll) {
    int b    = blockIdx.x * 8 + (threadIdx.x >> 5);
    int lane = threadIdx.x & 31;
    if (b >= BB) return;

    const float* P  = probs + (size_t)b * TT * KT;
    const int*   tg = label + (size_t)b * TT;
    int len = len_i[b];

    float tr[KT];
    #pragma unroll
    for (int j = 0; j < KT; ++j) tr[j] = (lane < KT) ? trans[j * KT + lane] : 0.0f;

    float alpha = (lane < KT) ? P[lane] : -1e30f;

    for (int t = 1; t < TT; ++t) {
        float pk = (lane < KT) ? P[t * KT + lane] : 0.0f;
        float vv[KT];
        float vmax = -1e30f;
        #pragma unroll
        for (int j = 0; j < KT; ++j) {
            float aj = __shfl(alpha, j, 32);
            vv[j] = aj + tr[j];
            vmax = fmaxf(vmax, vv[j]);
        }
        float s = 0.0f;
        #pragma unroll
        for (int j = 0; j < KT; ++j) s += __expf(vv[j] - vmax);
        float na = vmax + __logf(s) + pk;
        if (t < len) alpha = na;
    }

    // log-norm over the 10 tags
    float m2 = -1e30f;
    #pragma unroll
    for (int j = 0; j < KT; ++j) m2 = fmaxf(m2, __shfl(alpha, j, 32));
    float s2 = 0.0f;
    #pragma unroll
    for (int j = 0; j < KT; ++j) s2 += __expf(__shfl(alpha, j, 32) - m2);
    float ln = m2 + __logf(s2);

    // unary + binary scores, lanes stride over time
    float u = 0.0f, bs = 0.0f;
    for (int t = lane; t < TT; t += 32) {
        if (t < len) u += P[t * KT + tg[t]];
        if (t < len - 1) bs += trans[tg[t] * KT + tg[t + 1]];
    }
    #pragma unroll
    for (int off = 16; off > 0; off >>= 1) {
        u  += __shfl_xor(u, off, 32);
        bs += __shfl_xor(bs, off, 32);
    }
    if (lane == 0) out_ll[b] = u + bs - ((len == 0) ? 0.0f : ln);
}

// ---------------- launch ----------------
extern "C" void kernel_launch(void* const* d_in, const int* in_sizes, int n_in,
                              void* d_out, int out_size, void* d_ws, size_t ws_size,
                              hipStream_t stream) {
    const int*   word  = (const int*)d_in[0];
    const int*   label = (const int*)d_in[1];
    const float* emb   = (const float*)d_in[2];
    const float* Wf    = (const float*)d_in[3];
    const float* Uf    = (const float*)d_in[4];
    const float* bf    = (const float*)d_in[5];
    const float* Wb    = (const float*)d_in[6];
    const float* Ub    = (const float*)d_in[7];
    const float* bb    = (const float*)d_in[8];
    const float* Wd    = (const float*)d_in[9];
    const float* bd    = (const float*)d_in[10];
    const float* trans = (const float*)d_in[11];

    float* out = (float*)d_out;
    char*  ws  = (char*)d_ws;

    // workspace layout (~26.7 MB)
    _Float16* xh  = (_Float16*)(ws + 0);                      // 512*128*128 f16 = 16 MB
    _Float16* UtF = (_Float16*)(ws + 16777216);               // 1024*256 f16
    _Float16* UtB = (_Float16*)(ws + 17301504);
    _Float16* WtF = (_Float16*)(ws + 17825792);               // 1024*128 f16
    _Float16* WtB = (_Float16*)(ws + 18087936);
    float*    lgf = (float*)(ws + 18350080);                  // 512*128*16 f32
    float*    lgb = (float*)(ws + 22544384);
    int*      lni = (int*)(ws + 26738688);                    // 512 int

    float* out_len = out + (size_t)BB * TT * KT;              // 655360
    float* out_ll  = out_len + BB;                            // 655872

    // prep: embedding gather, lengths, weight transposes to f16
    k_embed<<<(BB * TT * EE) / 256, 256, 0, stream>>>(word, emb, xh);
    k_len<<<2, 256, 0, stream>>>(word, lni, out_len);
    k_t16<<<(EE * GG + 255) / 256, 256, 0, stream>>>(WtF, Wf, EE, GG);
    k_t16<<<(HH * GG + 255) / 256, 256, 0, stream>>>(UtF, Uf, HH, GG);
    k_t16<<<(EE * GG + 255) / 256, 256, 0, stream>>>(WtB, Wb, EE, GG);
    k_t16<<<(HH * GG + 255) / 256, 256, 0, stream>>>(UtB, Ub, HH, GG);

    // fused BiLSTM + partial dense (both directions concurrently)
    k_lstm<<<32, 512, 0, stream>>>(xh, WtF, UtF, WtB, UtB, bf, bb, Wd, lgf, lgb);

    // softmax -> probs
    k_soft<<<(BB * TT + 255) / 256, 256, 0, stream>>>(lgf, lgb, bd, out);

    // CRF log-likelihood
    k_crf<<<BB / 8, 256, 0, stream>>>(out, label, lni, trans, out_ll);
}
